// SNN6Class_33595234189636
// MI455X (gfx1250) — compile-verified, matrix operand order checked by reading, not run on hardware
//
#include <hip/hip_runtime.h>

// ---------------------------------------------------------------------------
// SNN 3-layer MLP, T=16 timesteps, B=16384, 512 -> 1024 -> 1024 -> 6
// fp16 WMMA (v_wmma_f32_16x16x32_f16) for all GEMMs, f32 accumulation,
// LIF neuron updates fused into GEMM epilogues.
// ---------------------------------------------------------------------------

typedef __attribute__((ext_vector_type(16))) _Float16     v16h;
typedef __attribute__((ext_vector_type(8)))  float        v8f;
typedef __attribute__((ext_vector_type(4)))  unsigned int u32x4;
typedef __attribute__((ext_vector_type(4)))  float        f32x4;

#define BSZ   16384
#define NIN   512
#define NHID  1024
#define NOUTP 16     // n_out=6 padded to 16 for one WMMA tile
#define NOUT  6
#define TSTEP 16

union FragU { u32x4 u[2]; v16h h; };

__device__ __forceinline__ v16h make_frag(const _Float16* p0, const _Float16* p1) {
    FragU f;
    f.u[0] = *reinterpret_cast<const u32x4*>(p0);
    f.u[1] = *reinterpret_cast<const u32x4*>(p1);
    return f.h;
}

// ---------------------------------------------------------------------------
// Utility kernels
// ---------------------------------------------------------------------------
__global__ void k_zero_f32(float* __restrict__ p, unsigned long long n) {
    unsigned long long i = (unsigned long long)blockIdx.x * blockDim.x + threadIdx.x;
    unsigned long long stride = (unsigned long long)gridDim.x * blockDim.x;
    for (; i < n; i += stride) p[i] = 0.0f;
}

__global__ void k_f32_to_f16(const float* __restrict__ src, _Float16* __restrict__ dst, int n) {
    int i = blockIdx.x * blockDim.x + threadIdx.x;
    if (i < n) dst[i] = (_Float16)src[i];
}

// W3 [6,1024] f32 -> W3h [16,1024] f16, rows 6..15 zero
__global__ void k_pad_w3(const float* __restrict__ W3, _Float16* __restrict__ W3h) {
    int i = blockIdx.x * blockDim.x + threadIdx.x;   // 0 .. 16*1024-1
    int row = i >> 10;
    int col = i & 1023;
    W3h[i] = (row < NOUT) ? (_Float16)W3[row * NHID + col] : (_Float16)0.0f;
}

// LIF layer 1: v1 = 0.5*(v1 + i1); s = (v1 >= 1); v1 *= (1-s)
__global__ void k_lif1(const _Float16* __restrict__ i1, float* __restrict__ v1,
                       _Float16* __restrict__ s1, int n) {
    int i = blockIdx.x * blockDim.x + threadIdx.x;
    if (i < n) {
        float v = 0.5f * (v1[i] + (float)i1[i]);
        float s = (v >= 1.0f) ? 1.0f : 0.0f;
        v1[i] = v * (1.0f - s);
        s1[i] = (_Float16)s;
    }
}

// ---------------------------------------------------------------------------
// fc1:  i1 = x @ W1^T + b1    (f32 x converted on the fly, fp16 WMMA)
// Grid: (NHID/256, BSZ/64), block 256 (8 waves, 2Mx4N wave grid, wave=32x64)
// ---------------------------------------------------------------------------
__global__ __launch_bounds__(256)
void k_fc1(const float* __restrict__ x, const _Float16* __restrict__ W1h,
           const float* __restrict__ b1, _Float16* __restrict__ i1) {
    __shared__ alignas(16) _Float16 aT[64 * 32];

    const int tid  = threadIdx.x;
    const int lane = tid & 31;
    const int w    = tid >> 5;
    const int wm   = w >> 2;          // 0..1
    const int wn   = w & 3;           // 0..3
    const int mWG  = blockIdx.y * 64;
    const int nWG  = blockIdx.x * 256;
    const int r    = lane & 15;       // A row / B col / D col
    const int kb   = (lane >> 4) * 8; // K-half select
    const int hi   = lane >> 4;

    const int srow = tid >> 2;        // 0..63  (staging row)
    const int sch  = (tid & 3) * 8;   // 0,8,16,24 (staging K chunk)

    v8f acc[2][4];
#pragma unroll
    for (int mi = 0; mi < 2; ++mi)
#pragma unroll
        for (int ni = 0; ni < 4; ++ni) acc[mi][ni] = {};

    for (int k0 = 0; k0 < NIN; k0 += 32) {
        __syncthreads();
        // stage A tile [64 x 32] with f32 -> f16 conversion
        {
            const float* xp = &x[(size_t)(mWG + srow) * NIN + k0 + sch];
            f32x4 f0 = *reinterpret_cast<const f32x4*>(xp);
            f32x4 f1 = *reinterpret_cast<const f32x4*>(xp + 4);
            _Float16* dst = &aT[srow * 32 + sch];
            dst[0] = (_Float16)f0.x; dst[1] = (_Float16)f0.y;
            dst[2] = (_Float16)f0.z; dst[3] = (_Float16)f0.w;
            dst[4] = (_Float16)f1.x; dst[5] = (_Float16)f1.y;
            dst[6] = (_Float16)f1.z; dst[7] = (_Float16)f1.w;
        }
        __syncthreads();

        v16h aF[2];
#pragma unroll
        for (int mi = 0; mi < 2; ++mi) {
            const _Float16* ap = &aT[(wm * 32 + mi * 16 + r) * 32];
            aF[mi] = make_frag(ap + kb, ap + 16 + kb);
        }
#pragma unroll
        for (int ni = 0; ni < 4; ++ni) {
            const int col = nWG + wn * 64 + ni * 16 + r;
            const _Float16* bp = &W1h[(size_t)col * NIN + k0];
            v16h bF = make_frag(bp + kb, bp + 16 + kb);
#pragma unroll
            for (int mi = 0; mi < 2; ++mi)
                acc[mi][ni] = __builtin_amdgcn_wmma_f32_16x16x32_f16(
                    false, aF[mi], false, bF, (short)0, acc[mi][ni], false, false);
        }
    }

    // epilogue: add bias, store i1 as fp16 (loop-invariant input current)
#pragma unroll
    for (int mi = 0; mi < 2; ++mi)
#pragma unroll
        for (int ni = 0; ni < 4; ++ni) {
            const int n = nWG + wn * 64 + ni * 16 + r;
            const float bias = b1[n];
#pragma unroll
            for (int e = 0; e < 8; ++e) {
                const int m = mWG + wm * 32 + mi * 16 + e + 8 * hi;
                i1[(size_t)m * NHID + n] = (_Float16)(acc[mi][ni][e] + bias);
            }
        }
}

// ---------------------------------------------------------------------------
// fc2 + LIF2 fused:  i2 = s1 @ W2^T + b2 ; v2 = 0.5*(v2+i2); s2 = spike(v2)
// Grid: (NHID/256, BSZ/64), block 256
// ---------------------------------------------------------------------------
__global__ __launch_bounds__(256)
void k_fc2_lif2(const _Float16* __restrict__ s1, const _Float16* __restrict__ W2h,
                const float* __restrict__ b2, float* __restrict__ v2,
                _Float16* __restrict__ s2) {
    __shared__ alignas(16) _Float16 aT[64 * 32];

    const int tid  = threadIdx.x;
    const int lane = tid & 31;
    const int w    = tid >> 5;
    const int wm   = w >> 2;
    const int wn   = w & 3;
    const int mWG  = blockIdx.y * 64;
    const int nWG  = blockIdx.x * 256;
    const int r    = lane & 15;
    const int kb   = (lane >> 4) * 8;
    const int hi   = lane >> 4;

    const int srow = tid >> 2;
    const int sch  = (tid & 3) * 8;

    v8f acc[2][4];
#pragma unroll
    for (int mi = 0; mi < 2; ++mi)
#pragma unroll
        for (int ni = 0; ni < 4; ++ni) acc[mi][ni] = {};

    for (int k0 = 0; k0 < NHID; k0 += 32) {
        __syncthreads();
        // stage s1 tile [64 x 32] (fp16, 16B per thread)
        *reinterpret_cast<u32x4*>(&aT[srow * 32 + sch]) =
            *reinterpret_cast<const u32x4*>(&s1[(size_t)(mWG + srow) * NHID + k0 + sch]);
        __syncthreads();

        v16h aF[2];
#pragma unroll
        for (int mi = 0; mi < 2; ++mi) {
            const _Float16* ap = &aT[(wm * 32 + mi * 16 + r) * 32];
            aF[mi] = make_frag(ap + kb, ap + 16 + kb);
        }
#pragma unroll
        for (int ni = 0; ni < 4; ++ni) {
            const int col = nWG + wn * 64 + ni * 16 + r;
            const _Float16* bp = &W2h[(size_t)col * NHID + k0];
            v16h bF = make_frag(bp + kb, bp + 16 + kb);
#pragma unroll
            for (int mi = 0; mi < 2; ++mi)
                acc[mi][ni] = __builtin_amdgcn_wmma_f32_16x16x32_f16(
                    false, aF[mi], false, bF, (short)0, acc[mi][ni], false, false);
        }
    }

    // fused LIF2 epilogue
#pragma unroll
    for (int mi = 0; mi < 2; ++mi)
#pragma unroll
        for (int ni = 0; ni < 4; ++ni) {
            const int n = nWG + wn * 64 + ni * 16 + r;
            const float bias = b2[n];
#pragma unroll
            for (int e = 0; e < 8; ++e) {
                const int m = mWG + wm * 32 + mi * 16 + e + 8 * hi;
                const size_t idx = (size_t)m * NHID + n;
                float v = 0.5f * (v2[idx] + acc[mi][ni][e] + bias);
                float s = (v >= 1.0f) ? 1.0f : 0.0f;
                v2[idx] = v * (1.0f - s);
                s2[idx] = (_Float16)s;
            }
        }
}

// ---------------------------------------------------------------------------
// fc3 + LIF3 + output accumulation:  out += spike(lif(s2 @ W3^T + b3)) / T
// One 16x16 WMMA tile per wave (cols 6..15 hit zero-padded W3 rows).
// Grid: BSZ/128 blocks of 256 (8 waves, 16 rows each)
// ---------------------------------------------------------------------------
__global__ __launch_bounds__(256)
void k_fc3_lif3(const _Float16* __restrict__ s2, const _Float16* __restrict__ W3h,
                const float* __restrict__ b3, float* __restrict__ v3,
                float* __restrict__ out) {
    const int tid   = threadIdx.x;
    const int lane  = tid & 31;
    const int w     = tid >> 5;
    const int mBase = blockIdx.x * 128 + w * 16;
    const int r     = lane & 15;
    const int kb    = (lane >> 4) * 8;
    const int hi    = lane >> 4;

    v8f acc = {};
    for (int k0 = 0; k0 < NHID; k0 += 32) {
        const _Float16* ap = &s2[(size_t)(mBase + r) * NHID + k0];
        v16h aF = make_frag(ap + kb, ap + 16 + kb);
        const _Float16* bp = &W3h[(size_t)r * NHID + k0];
        v16h bF = make_frag(bp + kb, bp + 16 + kb);
        acc = __builtin_amdgcn_wmma_f32_16x16x32_f16(
            false, aF, false, bF, (short)0, acc, false, false);
    }

    const int n = r;
    if (n < NOUT) {
        const float bias = b3[n];
#pragma unroll
        for (int e = 0; e < 8; ++e) {
            const int m = mBase + e + 8 * hi;
            const size_t vidx = (size_t)m * NOUTP + n;
            float v = 0.5f * (v3[vidx] + acc[e] + bias);
            float s = (v >= 1.0f) ? 1.0f : 0.0f;
            v3[vidx] = v * (1.0f - s);
            out[(size_t)m * NOUT + n] += s * (1.0f / (float)TSTEP);
        }
    }
}

// ---------------------------------------------------------------------------
// Host-side launcher
// ---------------------------------------------------------------------------
extern "C" void kernel_launch(void* const* d_in, const int* in_sizes, int n_in,
                              void* d_out, int out_size, void* d_ws, size_t ws_size,
                              hipStream_t stream) {
    const float* x  = (const float*)d_in[0];
    const float* W1 = (const float*)d_in[1];
    const float* b1 = (const float*)d_in[2];
    const float* W2 = (const float*)d_in[3];
    const float* b2 = (const float*)d_in[4];
    const float* W3 = (const float*)d_in[5];
    const float* b3 = (const float*)d_in[6];
    float* out = (float*)d_out;
    char*  ws  = (char*)d_ws;

    // workspace layout (all sizes multiples of 256B)
    constexpr size_t SZ_W1H = (size_t)NHID * NIN  * 2;   // 1 MB
    constexpr size_t SZ_W2H = (size_t)NHID * NHID * 2;   // 2 MB
    constexpr size_t SZ_W3H = (size_t)NOUTP * NHID * 2;  // 32 KB
    constexpr size_t SZ_I1  = (size_t)BSZ * NHID * 2;    // 32 MB
    constexpr size_t SZ_S   = (size_t)BSZ * NHID * 2;    // 32 MB
    constexpr size_t SZ_V   = (size_t)BSZ * NHID * 4;    // 64 MB

    size_t off = 0;
    _Float16* W1h = (_Float16*)(ws + off); off += SZ_W1H;
    _Float16* W2h = (_Float16*)(ws + off); off += SZ_W2H;
    _Float16* W3h = (_Float16*)(ws + off); off += SZ_W3H;
    _Float16* i1  = (_Float16*)(ws + off); off += SZ_I1;
    _Float16* s1  = (_Float16*)(ws + off); off += SZ_S;
    _Float16* s2  = (_Float16*)(ws + off); off += SZ_S;
    float*    v1  = (float*)   (ws + off); off += SZ_V;
    float*    v2  = (float*)   (ws + off); off += SZ_V;
    float*    v3  = (float*)   (ws + off); off += (size_t)BSZ * NOUTP * 4;

    // --- one-time setup: weight conversion, zero state ---
    k_f32_to_f16<<<(NHID * NIN) / 256, 256, 0, stream>>>(W1, W1h, NHID * NIN);
    k_f32_to_f16<<<(NHID * NHID) / 256, 256, 0, stream>>>(W2, W2h, NHID * NHID);
    k_pad_w3<<<(NOUTP * NHID) / 256, 256, 0, stream>>>(W3, W3h);

    k_zero_f32<<<4096, 256, 0, stream>>>(v1, (unsigned long long)BSZ * NHID);
    k_zero_f32<<<4096, 256, 0, stream>>>(v2, (unsigned long long)BSZ * NHID);
    k_zero_f32<<<1024, 256, 0, stream>>>(v3, (unsigned long long)BSZ * NOUTP);
    k_zero_f32<<<1024, 256, 0, stream>>>(out, (unsigned long long)BSZ * NOUT);

    // --- loop-invariant fc1 ---
    k_fc1<<<dim3(NHID / 256, BSZ / 64), 256, 0, stream>>>(x, W1h, b1, i1);

    // --- time loop ---
    for (int t = 0; t < TSTEP; ++t) {
        k_lif1<<<(BSZ * NHID) / 256, 256, 0, stream>>>(i1, v1, s1, BSZ * NHID);
        k_fc2_lif2<<<dim3(NHID / 256, BSZ / 64), 256, 0, stream>>>(s1, W2h, b2, v2, s2);
        k_fc3_lif3<<<BSZ / 128, 256, 0, stream>>>(s2, W3h, b3, v3, out);
    }
}
